// MultiHeadAttention_61332132987289
// MI455X (gfx1250) — matrix operand order repaired and checked by reading, not compiled, not run on hardware
//
#include <hip/hip_runtime.h>
#include <hip/hip_bf16.h>

typedef __bf16 bf16_t;
typedef __attribute__((ext_vector_type(16))) __bf16 v16bf;
typedef __attribute__((ext_vector_type(8)))  __bf16 v8bf;
typedef __attribute__((ext_vector_type(4)))  __bf16 v4bf;
typedef __attribute__((ext_vector_type(8)))  float  v8f;

#define D_MODEL   1024
#define NUM_HEADS 16
#define D_HEAD    64
#define SEQ       2048
#define BATCH     2
#define M_TOTAL   (BATCH * SEQ)      // 4096
#define ATTN_SCALE 0.125f            // 1/sqrt(64)

#if defined(__gfx1250__) && __has_builtin(__builtin_amdgcn_global_load_async_to_lds_b128)
#define HAVE_ASYNC_LDS 1
#else
#define HAVE_ASYNC_LDS 0
#endif

#if HAVE_ASYNC_LDS
typedef __attribute__((__vector_size__(4 * sizeof(int)))) int v4i_t;
typedef __attribute__((address_space(1))) v4i_t* gv4i_p;
typedef __attribute__((address_space(3))) v4i_t* lv4i_p;
#endif

// Copy 16 bytes global -> LDS. Async (ASYNCcnt) when available, else sync staging.
__device__ __forceinline__ void cp16_g2l(const bf16_t* g, bf16_t* l) {
#if HAVE_ASYNC_LDS
  __builtin_amdgcn_global_load_async_to_lds_b128((gv4i_p)g, (lv4i_p)l, 0, 0);
#else
  *(v8bf*)l = *(const v8bf*)g;
#endif
}

__device__ __forceinline__ void wait_staging() {
#if HAVE_ASYNC_LDS
  asm volatile("s_wait_asynccnt 0" ::: "memory");
#else
  asm volatile("s_wait_loadcnt 0\n\ts_wait_dscnt 0" ::: "memory");
#endif
}

// ---------------------------------------------------------------------------
// WMMA fragment loaders (wave32, 16x16x32 bf16).
// A 16x32 (MxK): lane = M row; halves 0..7 -> K = ks+0..7, halves 8..15 ->
// K = 16+ks+0..7, ks = 8*(lane>=16).
// ---------------------------------------------------------------------------
__device__ __forceinline__ v16bf load_frag_a(const bf16_t* base, int row, int ld, int k0) {
  const int lane = threadIdx.x & 31;
  const int r    = lane & 15;
  const int ks   = (lane >> 4) << 3;            // 0 or 8
  const bf16_t* p = base + (size_t)(row + r) * ld + k0 + ks;
  union { v16bf f; v8bf h[2]; } u;
  u.h[0] = *(const v8bf*)(p);
  u.h[1] = *(const v8bf*)(p + 16);
  return u.f;
}

// B 32x16 (KxN): lane = N column; halves 0..7 -> K = ks+0..7, halves 8..15 ->
// K = 16+ks+0..7, ks = 8*(lane>=16).
__device__ __forceinline__ v16bf load_frag_b(const bf16_t* base, int col0, int ld, int k0) {
  const int lane = threadIdx.x & 31;
  const int c    = lane & 15;
  const int ks   = (lane >> 4) << 3;            // 0 or 8
  const bf16_t* p = base + (size_t)(col0 + c) * ld + k0 + ks;
  union { v16bf f; v8bf h[2]; } u;
  u.h[0] = *(const v8bf*)(p);
  u.h[1] = *(const v8bf*)(p + 16);
  return u.f;
}

__device__ __forceinline__ v8f wmma_bf16(v16bf a, v16bf b, v8f c) {
  return __builtin_amdgcn_wmma_f32_16x16x32_bf16(false, a, false, b, (short)0, c,
                                                 false, false);
}

// ---------------------------------------------------------------------------
// fp32 -> bf16 bulk convert (vectorized x4)
// ---------------------------------------------------------------------------
__global__ __launch_bounds__(256) void cvt_bf16(const float* __restrict__ in,
                                                bf16_t* __restrict__ out, int n4) {
  int i = blockIdx.x * blockDim.x + threadIdx.x;
  if (i < n4) {
    float4 v = ((const float4*)in)[i];
    v4bf o;
    o[0] = (bf16_t)v.x; o[1] = (bf16_t)v.y; o[2] = (bf16_t)v.z; o[3] = (bf16_t)v.w;
    ((v4bf*)out)[i] = o;
  }
}

// ---------------------------------------------------------------------------
// GEMM: C[m,n] = scale * (sum_k A[m,k]*W[n,k] + bias[n])
// A: [M_TOTAL, K] bf16; W: [N, K] bf16 (torch Linear layout).
// LDS-staged, double buffered via async-to-LDS copies.
// Block: 256 thr = 8 waves as 2(M)x4(N); wave 32x64 -> block 64x256.
// MODE 0: bf16 out [B,H,S,Dh] (Q,K)  MODE 1: bf16 out [B,H,Dh,S] (V)
// MODE 2: fp32 out [M, D_MODEL]      (output projection)
// ---------------------------------------------------------------------------
template <int MODE>
__global__ __launch_bounds__(256)
void gemm_nt(const bf16_t* __restrict__ A, const bf16_t* __restrict__ W,
             const float* __restrict__ bias, void* __restrict__ Cout,
             int K, float scale) {
  __shared__ __align__(16) bf16_t lA[2][64 * 32];    //  4 KB per buffer
  __shared__ __align__(16) bf16_t lW[2][256 * 32];   // 16 KB per buffer

  const int tid  = threadIdx.x;
  const int wid  = tid >> 5;
  const int lane = tid & 31;
  const int m0b = blockIdx.x * 64;
  const int n0b = blockIdx.y * 256;
  const int wm  = (wid >> 2) * 32;
  const int wn  = (wid & 3) * 64;

  // staging assignments (16B chunks)
  const int arow = tid >> 2, acol = (tid & 3) * 8;   // A: 64x32 = 256 chunks

  v8f acc[2][4];
#pragma unroll
  for (int mi = 0; mi < 2; ++mi)
#pragma unroll
    for (int nt = 0; nt < 4; ++nt) acc[mi][nt] = (v8f){0.f,0.f,0.f,0.f,0.f,0.f,0.f,0.f};

  auto prefetch = [&](int buf, int k) {
    cp16_g2l(A + (size_t)(m0b + arow) * K + k + acol, &lA[buf][arow * 32 + acol]);
#pragma unroll
    for (int i = 0; i < 4; ++i) {                    // W: 256x32 = 1024 chunks
      const int ch = tid + 256 * i;
      const int r = ch >> 2, c8 = (ch & 3) * 8;
      cp16_g2l(W + (size_t)(n0b + r) * K + k + c8, &lW[buf][r * 32 + c8]);
    }
  };

  prefetch(0, 0);
  int buf = 0;
#pragma unroll 1
  for (int k = 0; k < K; k += 32) {
    wait_staging();
    __syncthreads();
    if (k + 32 < K) prefetch(buf ^ 1, k + 32);

    v16bf a0 = load_frag_a(lA[buf], wm,      32, 0);
    v16bf a1 = load_frag_a(lA[buf], wm + 16, 32, 0);
#pragma unroll
    for (int nt = 0; nt < 4; ++nt) {
      v16bf bf = load_frag_b(lW[buf], wn + 16 * nt, 32, 0);
      acc[0][nt] = wmma_bf16(a0, bf, acc[0][nt]);
      acc[1][nt] = wmma_bf16(a1, bf, acc[1][nt]);
    }
    buf ^= 1;
  }

  // C/D layout: lane -> col = lane&15 (+16*nt), VGPR j -> row j + 8*(lane>=16)
  const int c  = lane & 15;
  const int rb = (lane >> 4) << 3;
  const int m0 = m0b + wm, n0 = n0b + wn;
#pragma unroll
  for (int mi = 0; mi < 2; ++mi) {
#pragma unroll
    for (int nt = 0; nt < 4; ++nt) {
      const int n  = n0 + 16 * nt + c;
      const float bv = bias[n];
#pragma unroll
      for (int j = 0; j < 8; ++j) {
        const int m = m0 + 16 * mi + rb + j;
        const float val = (acc[mi][nt][j] + bv) * scale;
        if (MODE == 2) {
          ((float*)Cout)[(size_t)m * D_MODEL + n] = val;
        } else {
          const int b = m >> 11, s = m & (SEQ - 1);
          const int h = n >> 6,  d = n & (D_HEAD - 1);
          bf16_t* o = (bf16_t*)Cout;
          if (MODE == 0)
            o[(((size_t)(b * NUM_HEADS + h)) * SEQ + s) * D_HEAD + d] = (bf16_t)val;
          else
            o[(((size_t)(b * NUM_HEADS + h)) * D_HEAD + d) * SEQ + s] = (bf16_t)val;
        }
      }
    }
  }
}

// ---------------------------------------------------------------------------
// Flash attention: grid (SEQ/256, B*H), block 256 (8 waves).
// K/V tiles for the 32-key block are staged once per block in LDS (async,
// double buffered) and shared by all 8 waves; each wave owns 32 q-rows.
// Scale folded into Q. Output: merged bf16 [B*S, D_MODEL].
// ---------------------------------------------------------------------------
__global__ __launch_bounds__(256)
void attn_kernel(const bf16_t* __restrict__ Qh, const bf16_t* __restrict__ Kh,
                 const bf16_t* __restrict__ Vt, bf16_t* __restrict__ AO) {
  __shared__ __align__(16) bf16_t lK[2][32 * 64];     // 4 KB per buffer
  __shared__ __align__(16) bf16_t lV[2][64 * 32];     // 4 KB per buffer
  __shared__ __align__(16) bf16_t plds[8][32 * 32];   // per-wave P tile, 16 KB

  const int tid  = threadIdx.x;
  const int wid  = tid >> 5;
  const int lane = tid & 31;
  const int bh = blockIdx.y;
  const int b = bh >> 4, h = bh & 15;
  const int q0 = blockIdx.x * 256 + wid * 32;

  const bf16_t* Q  = Qh + (size_t)bh * SEQ * D_HEAD;
  const bf16_t* Kp = Kh + (size_t)bh * SEQ * D_HEAD;
  const bf16_t* Vp = Vt + (size_t)bh * D_HEAD * SEQ;

  // staging assignments (16B chunks)
  const int krow = tid >> 3, kcol = (tid & 7) * 8;   // K tile 32x64: 256 chunks
  const int vrow = tid >> 2, vcol = (tid & 3) * 8;   // V tile 64x32: 256 chunks

  auto prefetch = [&](int buf, int kb) {
    cp16_g2l(Kp + (size_t)(kb + krow) * D_HEAD + kcol, &lK[buf][krow * 64 + kcol]);
    cp16_g2l(Vp + (size_t)vrow * SEQ + kb + vcol,      &lV[buf][vrow * 32 + vcol]);
  };

  v16bf qf[2][2];
#pragma unroll
  for (int mi = 0; mi < 2; ++mi)
#pragma unroll
    for (int kc = 0; kc < 2; ++kc)
      qf[mi][kc] = load_frag_a(Q, q0 + 16 * mi, D_HEAD, 32 * kc);

  v8f o[2][4];
  float mo[2][8], li[2][8];
#pragma unroll
  for (int mi = 0; mi < 2; ++mi) {
#pragma unroll
    for (int nt = 0; nt < 4; ++nt) o[mi][nt] = (v8f){0.f,0.f,0.f,0.f,0.f,0.f,0.f,0.f};
#pragma unroll
    for (int j = 0; j < 8; ++j) { mo[mi][j] = -3.0e38f; li[mi][j] = 0.f; }
  }

  const int c  = lane & 15;
  const int rb = (lane >> 4) << 3;
  bf16_t* myp = plds[wid];

  prefetch(0, 0);
  int buf = 0;
#pragma unroll 1
  for (int kb = 0; kb < SEQ; kb += 32) {
    wait_staging();
    __syncthreads();
    if (kb + 32 < SEQ) prefetch(buf ^ 1, kb + 32);

    // ---- scores: S = Q K^T  (32q x 32k), Dh=64 -> 2 chained K=32 WMMAs ----
    v8f s[2][2];
#pragma unroll
    for (int mi = 0; mi < 2; ++mi)
#pragma unroll
      for (int nt = 0; nt < 2; ++nt) {
        v8f z = (v8f){0.f,0.f,0.f,0.f,0.f,0.f,0.f,0.f};
        z = wmma_bf16(qf[mi][0], load_frag_b(lK[buf], 16 * nt, 64, 0),  z);
        s[mi][nt] = wmma_bf16(qf[mi][1], load_frag_b(lK[buf], 16 * nt, 64, 32), z);
      }

    // ---- online softmax (rows live across 16 lanes of a half-wave) ----
#pragma unroll
    for (int mi = 0; mi < 2; ++mi) {
      float rm[8];
#pragma unroll
      for (int j = 0; j < 8; ++j) rm[j] = fmaxf(s[mi][0][j], s[mi][1][j]);
#pragma unroll
      for (int mask = 1; mask <= 8; mask <<= 1)
#pragma unroll
        for (int j = 0; j < 8; ++j) rm[j] = fmaxf(rm[j], __shfl_xor(rm[j], mask, 32));

      float alpha[8];
#pragma unroll
      for (int j = 0; j < 8; ++j) {
        const float mn = fmaxf(mo[mi][j], rm[j]);
        alpha[j] = __expf(mo[mi][j] - mn);
        mo[mi][j] = mn;
      }
#pragma unroll
      for (int nt = 0; nt < 2; ++nt)
#pragma unroll
        for (int j = 0; j < 8; ++j) s[mi][nt][j] = __expf(s[mi][nt][j] - mo[mi][j]);

      float rs[8];
#pragma unroll
      for (int j = 0; j < 8; ++j) rs[j] = s[mi][0][j] + s[mi][1][j];
#pragma unroll
      for (int mask = 1; mask <= 8; mask <<= 1)
#pragma unroll
        for (int j = 0; j < 8; ++j) rs[j] += __shfl_xor(rs[j], mask, 32);
#pragma unroll
      for (int j = 0; j < 8; ++j) li[mi][j] = li[mi][j] * alpha[j] + rs[j];
#pragma unroll
      for (int nt = 0; nt < 4; ++nt)
#pragma unroll
        for (int j = 0; j < 8; ++j) o[mi][nt][j] *= alpha[j];

      // stash P tile (bf16, row-major 32x32) into this wave's LDS region
#pragma unroll
      for (int nt = 0; nt < 2; ++nt)
#pragma unroll
        for (int j = 0; j < 8; ++j)
          myp[(16 * mi + rb + j) * 32 + 16 * nt + c] = (bf16_t)s[mi][nt][j];
    }

    asm volatile("s_wait_dscnt 0" ::: "memory");   // DS ops are per-wave in-order

    // ---- O += P V  (P: 32x32 A-frags from LDS, V tile staged in LDS) ----
    v16bf pa0 = load_frag_a(myp, 0,  32, 0);
    v16bf pa1 = load_frag_a(myp, 16, 32, 0);
#pragma unroll
    for (int nt = 0; nt < 4; ++nt) {
      v16bf vb = load_frag_b(lV[buf], 16 * nt, 32, 0);
      o[0][nt] = wmma_bf16(pa0, vb, o[0][nt]);
      o[1][nt] = wmma_bf16(pa1, vb, o[1][nt]);
    }
    buf ^= 1;
  }

  // ---- normalize, write merged bf16 [B*S, D_MODEL] ----
#pragma unroll
  for (int mi = 0; mi < 2; ++mi)
#pragma unroll
    for (int j = 0; j < 8; ++j) {
      const float inv = 1.0f / li[mi][j];
      const int m = q0 + 16 * mi + rb + j;
      const size_t row = ((size_t)b * SEQ + m) * D_MODEL + h * D_HEAD;
#pragma unroll
      for (int nt = 0; nt < 4; ++nt)
        AO[row + 16 * nt + c] = (bf16_t)(o[mi][nt][j] * inv);
    }
}

// ---------------------------------------------------------------------------
extern "C" void kernel_launch(void* const* d_in, const int* in_sizes, int n_in,
                              void* d_out, int out_size, void* d_ws, size_t ws_size,
                              hipStream_t stream) {
  const float* X  = (const float*)d_in[0];
  const float* Wq = (const float*)d_in[1];
  const float* bq = (const float*)d_in[2];
  const float* Wk = (const float*)d_in[3];
  const float* bk = (const float*)d_in[4];
  const float* Wv = (const float*)d_in[5];
  const float* bv = (const float*)d_in[6];
  const float* Wo = (const float*)d_in[7];
  const float* bo = (const float*)d_in[8];

  char* ws = (char*)d_ws;
  const size_t MB = 1u << 20;
  bf16_t* Xb  = (bf16_t*)(ws + 0 * MB);    // 4096x1024  (8 MB)
  bf16_t* Wqb = (bf16_t*)(ws + 8 * MB);    // 1024x1024  (2 MB each)
  bf16_t* Wkb = (bf16_t*)(ws + 10 * MB);
  bf16_t* Wvb = (bf16_t*)(ws + 12 * MB);
  bf16_t* Wob = (bf16_t*)(ws + 14 * MB);
  bf16_t* Qh  = (bf16_t*)(ws + 16 * MB);   // [B,H,S,Dh] (8 MB)
  bf16_t* Kh  = (bf16_t*)(ws + 24 * MB);   // [B,H,S,Dh]
  bf16_t* Vt  = (bf16_t*)(ws + 32 * MB);   // [B,H,Dh,S]
  bf16_t* AO  = (bf16_t*)(ws + 40 * MB);   // [B*S, D_MODEL]

  const int nX4 = M_TOTAL * D_MODEL / 4;
  const int nW4 = D_MODEL * D_MODEL / 4;
  cvt_bf16<<<nX4 / 256, 256, 0, stream>>>(X,  Xb,  nX4);
  cvt_bf16<<<nW4 / 256, 256, 0, stream>>>(Wq, Wqb, nW4);
  cvt_bf16<<<nW4 / 256, 256, 0, stream>>>(Wk, Wkb, nW4);
  cvt_bf16<<<nW4 / 256, 256, 0, stream>>>(Wv, Wvb, nW4);
  cvt_bf16<<<nW4 / 256, 256, 0, stream>>>(Wo, Wob, nW4);

  dim3 g(M_TOTAL / 64, D_MODEL / 256), blk(256);
  gemm_nt<0><<<g, blk, 0, stream>>>(Xb, Wqb, bq, Qh, D_MODEL, ATTN_SCALE);
  gemm_nt<0><<<g, blk, 0, stream>>>(Xb, Wkb, bk, Kh, D_MODEL, 1.0f);
  gemm_nt<1><<<g, blk, 0, stream>>>(Xb, Wvb, bv, Vt, D_MODEL, 1.0f);

  attn_kernel<<<dim3(SEQ / 256, BATCH * NUM_HEADS), blk, 0, stream>>>(Qh, Kh, Vt, AO);

  gemm_nt<2><<<g, blk, 0, stream>>>(AO, Wob, bo, d_out, D_MODEL, 1.0f);
}
